// InteractionModule_49632642072858
// MI455X (gfx1250) — matrix operand order repaired and checked by reading, not compiled, not run on hardware
//
#include <hip/hip_runtime.h>

typedef __attribute__((ext_vector_type(16))) _Float16 v16h;
typedef __attribute__((ext_vector_type(8)))  _Float16 v8h;
typedef __attribute__((ext_vector_type(8)))  float    v8f;

#define SQ3 1.7320508075688772f
#define SQ2 1.4142135623730951f
#define ISQRTD 5.0f   /* float(math.isqrt(32)) */

// ---- optional gfx1250 async global->LDS path ----------------------------
#if defined(__has_builtin)
#if __has_builtin(__builtin_amdgcn_global_load_async_to_lds_b128) && \
    __has_builtin(__builtin_amdgcn_s_wait_asynccnt)
#define HAVE_ASYNC_LDS 1
#endif
#endif

#if defined(HAVE_ASYNC_LDS)
typedef int b128_t __attribute__((vector_size(16)));
typedef __attribute__((address_space(1))) b128_t* gbl_b128_p;
typedef __attribute__((address_space(3))) b128_t* lds_b128_p;
#endif

__device__ __forceinline__ void async_copy16(void* lds, const void* g) {
#if defined(HAVE_ASYNC_LDS)
  __builtin_amdgcn_global_load_async_to_lds_b128(
      (gbl_b128_p)(unsigned long long)g,
      (lds_b128_p)(unsigned int)(unsigned long long)lds, 0, 0);
#else
  *(float4*)lds = *(const float4*)g;
#endif
}
__device__ __forceinline__ void async_join() {
#if defined(HAVE_ASYNC_LDS)
  __builtin_amdgcn_s_wait_asynccnt(0);
#endif
}

// ---- WMMA fragment builders (16x16x32 f16) ------------------------------
// A (16xK tile row-major, rowp = start of this lane's M row):
//   lane<16 : K = k0+0..15 ; lane>=16 : K = k0+16..31
__device__ __forceinline__ v16h a_frag(const _Float16* rowp, int k0, int g) {
  const v8h lo = *(const v8h*)(rowp + k0 + 16 * g);
  const v8h hi = *(const v8h*)(rowp + k0 + 16 * g + 8);
  v16h r;
#pragma unroll
  for (int i = 0; i < 8; ++i) { r[i] = lo[i]; r[8 + i] = hi[i]; }
  return r;
}
// B from transposed weights Wt[n][k] (contiguous K): lane needs K = k0+16g .. +15
__device__ __forceinline__ v16h b_frag(const _Float16* p) {
  const v8h lo = *(const v8h*)(p);
  const v8h hi = *(const v8h*)(p + 8);
  v16h r;
#pragma unroll
  for (int i = 0; i < 8; ++i) { r[i] = lo[i]; r[8 + i] = hi[i]; }
  return r;
}
#define WMMA(a, b, c) \
  __builtin_amdgcn_wmma_f32_16x16x32_f16(false, (a), false, (b), (short)0, (c), false, false)

// ---- weight prep --------------------------------------------------------
__global__ void k_w2h(const float* __restrict__ src, _Float16* __restrict__ dst,
                      int K, int N, int Kpad) {
  int idx = blockIdx.x * blockDim.x + threadIdx.x;
  if (idx >= N * Kpad) return;
  int n = idx / Kpad, k = idx % Kpad;
  dst[idx] = (k < K) ? (_Float16)src[(size_t)k * N + n] : (_Float16)0.f;
}
__global__ void k_w2f(const float* __restrict__ src, float* __restrict__ dst,
                      int K, int C) {
  int idx = blockIdx.x * blockDim.x + threadIdx.x;
  if (idx >= C * K) return;
  int c = idx / K, k = idx % K;
  dst[idx] = src[(size_t)k * C + c];
}

__global__ void k_copy(const float* __restrict__ s, float* __restrict__ d, int n) {
  int i = blockIdx.x * blockDim.x + threadIdx.x;
  if (i < n) d[i] = s[i];
}

__global__ void k_init(float* mx, float* den, float* counts, float* local_env, int N) {
  int i = blockIdx.x * blockDim.x + threadIdx.x;
  if (i < N * 32) { mx[i] = -3.402823466e38f; den[i] = 0.f; }
  if (i < N) counts[i] = 0.f;
  if (i < N * 128) local_env[i] = 0.f;
}

// ---- K1: LN -> GEMM1 -> SiLU -> GEMM2 -> cutoff -> scatter latents ------
// 64 edges / block, 8 waves; B fragments cached in registers across 4 row-tiles.
__global__ void __launch_bounds__(256) k_mlp(
    const float* __restrict__ xin, const float* __restrict__ ln_g,
    const float* __restrict__ ln_b, const _Float16* __restrict__ W1t,
    const float* __restrict__ b1, const _Float16* __restrict__ W2t,
    const float* __restrict__ b2, const float* __restrict__ cutoff,
    const int* __restrict__ aedges, float* __restrict__ latents_out, int E) {
  __shared__ __align__(16) _Float16 Apan[64][168];   // 64 edges x 160 (pad)
  __shared__ __align__(16) _Float16 Hpan[64][264];   // 64 edges x 256 (pad)
  const int tid = threadIdx.x, wave = tid >> 5, lane = tid & 31;
  const int g = lane >> 4, nrow = lane & 15;
  const int e0 = blockIdx.x * 64;

  if (tid < 64) {                       // layer norm, one row per thread
    int r = tid;
    int e = e0 + r; if (e >= E) e = E - 1;
    const float* x = xin + (size_t)e * 136;
    float mu = 0.f;
    for (int k = 0; k < 136; ++k) mu += x[k];
    mu *= (1.f / 136.f);
    float var = 0.f;
    for (int k = 0; k < 136; ++k) { float d = x[k] - mu; var += d * d; }
    var *= (1.f / 136.f);
    float rs = rsqrtf(var + 1e-5f);
    for (int k = 0; k < 136; ++k)
      Apan[r][k] = (_Float16)((x[k] - mu) * rs * ln_g[k] + ln_b[k]);
    for (int k = 136; k < 160; ++k) Apan[r][k] = (_Float16)0.f;
  }
  __builtin_prefetch(W1t + (size_t)(wave * 2 * 16) * 160, 0, 3);
  __syncthreads();

  // h = silu(A@W1 + b1): 256 cols -> 16 col-tiles, 2 per wave, x4 row-tiles
  for (int tt = 0; tt < 2; ++tt) {
    const int ct = wave * 2 + tt;
    const int col = ct * 16 + nrow;
    v16h Bc[5];
#pragma unroll
    for (int ks = 0; ks < 5; ++ks)
      Bc[ks] = b_frag(W1t + (size_t)col * 160 + ks * 32 + 16 * g);
    const float bias = b1[col];
    for (int rt = 0; rt < 4; ++rt) {
      v8f acc = {};
#pragma unroll
      for (int ks = 0; ks < 5; ++ks)
        acc = WMMA(a_frag(&Apan[rt * 16 + nrow][0], ks * 32, g), Bc[ks], acc);
#pragma unroll
      for (int i = 0; i < 8; ++i) {
        float v = acc[i] + bias;
        float s = v * __builtin_amdgcn_rcpf(1.f + __expf(-v));
        Hpan[rt * 16 + i + 8 * g][col] = (_Float16)s;
      }
    }
  }
  __syncthreads();

  // latents = cutoff * (h@W2 + b2): 128 cols -> 8 col-tiles, 1 per wave
  {
    const int ct = wave;
    const int col = ct * 16 + nrow;
    v16h Bc[8];
#pragma unroll
    for (int ks = 0; ks < 8; ++ks)
      Bc[ks] = b_frag(W2t + (size_t)col * 256 + ks * 32 + 16 * g);
    const float bias = b2[col];
    for (int rt = 0; rt < 4; ++rt) {
      v8f acc = {};
#pragma unroll
      for (int ks = 0; ks < 8; ++ks)
        acc = WMMA(a_frag(&Hpan[rt * 16 + nrow][0], ks * 32, g), Bc[ks], acc);
#pragma unroll
      for (int i = 0; i < 8; ++i) {
        int e = e0 + rt * 16 + i + 8 * g;
        if (e < E) {
          int ae = aedges[e];
          latents_out[(size_t)ae * 128 + col] = cutoff[e] * (acc[i] + bias);
        }
      }
    }
  }
}

// ---- K2: Q/K via WMMA, logits -------------------------------------------
// 64 edges / block, 8 waves; each wave owns 4 heads; Bq/Bk cached across rt.
__global__ void __launch_bounds__(256) k_qk(
    const float* __restrict__ node_inv, const float* __restrict__ edge_inv,
    const float* __restrict__ latents, const _Float16* __restrict__ Wqt,
    const _Float16* __restrict__ Wkt, const int* __restrict__ ecen,
    const int* __restrict__ enb, float* __restrict__ logits, int E) {
  __shared__ __align__(16) _Float16 Fpan[64][168];  // edge_full_attr (160 pad)
  __shared__ __align__(16) _Float16 Lpan[64][136];  // latents (128)
  const int tid = threadIdx.x;
  const int e0 = blockIdx.x * 64;
  for (int idx = tid; idx < 64 * 160; idx += 256) {
    int r = idx / 160, k = idx % 160;
    int e = e0 + r; if (e >= E) e = E - 1;
    float v;
    if (k < 64)       v = node_inv[(size_t)ecen[e] * 64 + k];
    else if (k < 128) v = node_inv[(size_t)enb[e] * 64 + (k - 64)];
    else if (k < 136) v = edge_inv[(size_t)e * 8 + (k - 128)];
    else              v = 0.f;
    Fpan[r][k] = (_Float16)v;
  }
  for (int idx = tid; idx < 64 * 128; idx += 256) {
    int r = idx / 128, k = idx % 128;
    int e = e0 + r; if (e >= E) e = E - 1;
    Lpan[r][k] = (_Float16)latents[(size_t)e * 128 + k];
  }
  __syncthreads();
  const int wave = tid >> 5, lane = tid & 31, g = lane >> 4, n = lane & 15;
  for (int j = 0; j < 4; ++j) {     // 4 heads per wave (M=32 heads)
    const int m = wave * 4 + j;
    float lacc[4][8];
#pragma unroll
    for (int rt = 0; rt < 4; ++rt)
#pragma unroll
      for (int i = 0; i < 8; ++i) lacc[rt][i] = 0.f;

    for (int t = 0; t < 2; ++t) {   // head m spans col-tiles 2m, 2m+1
      const int ct = 2 * m + t;
      v16h Bq[5], Bk[4];
#pragma unroll
      for (int ks = 0; ks < 5; ++ks)
        Bq[ks] = b_frag(Wqt + (size_t)(ct * 16 + n) * 160 + ks * 32 + 16 * g);
#pragma unroll
      for (int ks = 0; ks < 4; ++ks)
        Bk[ks] = b_frag(Wkt + (size_t)(ct * 16 + n) * 128 + ks * 32 + 16 * g);
#pragma unroll
      for (int rt = 0; rt < 4; ++rt) {
        v8f q = {};
#pragma unroll
        for (int ks = 0; ks < 5; ++ks)
          q = WMMA(a_frag(&Fpan[rt * 16 + n][0], ks * 32, g), Bq[ks], q);
        v8f kk = {};
#pragma unroll
        for (int ks = 0; ks < 4; ++ks)
          kk = WMMA(a_frag(&Lpan[rt * 16 + n][0], ks * 32, g), Bk[ks], kk);
#pragma unroll
        for (int i = 0; i < 8; ++i) {   // dot over 16 dh cols in this tile
          float p = q[i] * kk[i];
          p += __shfl_xor(p, 1, 32);
          p += __shfl_xor(p, 2, 32);
          p += __shfl_xor(p, 4, 32);
          p += __shfl_xor(p, 8, 32);
          lacc[rt][i] += p;
        }
      }
    }
    if (n == 0) {
#pragma unroll
      for (int rt = 0; rt < 4; ++rt)
#pragma unroll
        for (int i = 0; i < 8; ++i) {
          int e = e0 + rt * 16 + i + 8 * g;
          if (e < E) logits[(size_t)e * 32 + m] = lacc[rt][i] * ISQRTD;
        }
    }
  }
}

// ---- segment softmax ----------------------------------------------------
__device__ __forceinline__ void atomicMaxF(float* addr, float val) {
  if (val >= 0.f) atomicMax((int*)addr, __float_as_int(val));
  else            atomicMin((unsigned int*)addr, __float_as_uint(val));
}

__global__ void k_segmax(const float* __restrict__ logits,
                         const int* __restrict__ ecen, float* mx, int EM) {
  int i = blockIdx.x * blockDim.x + threadIdx.x;
  if (i >= EM) return;
  int e = i >> 5, m = i & 31;
  atomicMaxF(&mx[(size_t)ecen[e] * 32 + m], logits[i]);
}

__global__ void k_exp(float* __restrict__ logits, const int* __restrict__ ecen,
                      const float* __restrict__ mx, float* den, int EM) {
  int i = blockIdx.x * blockDim.x + threadIdx.x;
  if (i >= EM) return;
  int e = i >> 5, m = i & 31;
  int c = ecen[e];
  float ex = __expf(logits[i] - mx[(size_t)c * 32 + m]);
  logits[i] = ex;
  atomicAdd(&den[(size_t)c * 32 + m], ex);
}

// ---- attention-weighted local environment scatter -----------------------
__global__ void __launch_bounds__(256) k_env(
    const float* __restrict__ latents, const float* __restrict__ Wenvt,
    const float* __restrict__ edge_attr, const float* __restrict__ exbuf,
    const float* __restrict__ den, const int* __restrict__ ecen,
    float* local_env, float* counts, int E) {
  __shared__ __align__(16) float lat[8][128];
  __shared__ __align__(16) float Wsh[64][128];   // W_env emb half (cols 64..127)
  const int tid = threadIdx.x;
  {
    const float4* src = (const float4*)(Wenvt + (size_t)64 * 128);
    float4* dst = (float4*)&Wsh[0][0];
    for (int i = tid; i < 64 * 128 / 4; i += 256) async_copy16(dst + i, src + i);
  }
  const int wave = tid >> 5, lane = tid & 31;
  const int e = blockIdx.x * 8 + wave;
  const bool ok = e < E;
  if (ok)
    for (int k = lane; k < 128; k += 32) lat[wave][k] = latents[(size_t)e * 128 + k];
  async_join();
  __syncthreads();
  if (!ok) return;
  const int c = ecen[e], m = lane;
  const float4* lw0 = (const float4*)&Wsh[2 * m][0];
  const float4* lw1 = (const float4*)&Wsh[2 * m + 1][0];
  const float4* ll  = (const float4*)&lat[wave][0];
  float w0 = 0.f, w1 = 0.f;
#pragma unroll 8
  for (int k = 0; k < 32; ++k) {
    float4 l = ll[k], a = lw0[k], b = lw1[k];
    w0 += l.x * a.x + l.y * a.y + l.z * a.z + l.w * a.w;
    w1 += l.x * b.x + l.y * b.y + l.z * b.z + l.w * b.w;
  }
  float attn = exbuf[(size_t)e * 32 + m] / den[(size_t)c * 32 + m];
  float a0 = edge_attr[(size_t)e * 4 + 0], a1 = edge_attr[(size_t)e * 4 + 1];
  float a2 = edge_attr[(size_t)e * 4 + 2], a3 = edge_attr[(size_t)e * 4 + 3];
  float* d = local_env + ((size_t)c * 32 + m) * 4;
  atomicAdd(d + 0, a0 * w0 * attn);
  atomicAdd(d + 1, a1 * w1 * attn);
  atomicAdd(d + 2, a2 * w1 * attn);
  atomicAdd(d + 3, a3 * w1 * attn);
  if (lane == 0) atomicAdd(&counts[c], 1.f);
}

// ---- per-node SO(3) LN + linears ---------------------------------------
__global__ void __launch_bounds__(256) k_node(
    const float* __restrict__ local_env, const float* __restrict__ counts,
    const float* __restrict__ W_el0, const float* __restrict__ W_el1,
    float* env_node, int N) {
  __shared__ float le[8][128];
  const int wave = threadIdx.x >> 5, lane = threadIdx.x & 31;
  const int nId = blockIdx.x * 8 + wave;
  const bool ok = nId < N;
  if (ok)
    for (int k = lane; k < 128; k += 32) le[wave][k] = local_env[(size_t)nId * 128 + k];
  __syncthreads();
  if (!ok) return;
  float s = le[wave][lane * 4];
  float s2 = s * s, v2 = 0.f;
  for (int cc = 0; cc < 3; ++cc) { float v = le[wave][lane * 4 + 1 + cc]; v2 += v * v; }
  for (int msk = 16; msk >= 1; msk >>= 1) {
    s2 += __shfl_xor(s2, msk, 32);
    v2 += __shfl_xor(v2, msk, 32);
  }
  float ir0 = 1.f / sqrtf(s2 * (1.f / 32.f) + 1e-5f);
  float ir1 = 1.f / sqrtf(v2 * (1.f / 96.f) + 1e-5f);
  const int m = lane;
  float ls = 0.f, lv0 = 0.f, lv1 = 0.f, lv2 = 0.f;
  for (int u = 0; u < 32; ++u) {
    float w0 = W_el0[u * 32 + m], w1 = W_el1[u * 32 + m];
    ls  += le[wave][u * 4 + 0] * w0;
    lv0 += le[wave][u * 4 + 1] * w1;
    lv1 += le[wave][u * 4 + 2] * w1;
    lv2 += le[wave][u * 4 + 3] * w1;
  }
  float sc = counts[nId] > 0.f ? 1.f : 0.f;
  float* d = env_node + (size_t)nId * 128 + m * 4;
  d[0] = sc * ls * ir0;
  d[1] = sc * lv0 * ir1;
  d[2] = sc * lv1 * ir1;
  d[3] = sc * lv2 * ir1;
}

// ---- final: CG tensor product + SO(3) LN + output linears ---------------
__global__ void __launch_bounds__(256) k_final(
    const float* __restrict__ latents, const float* __restrict__ Wenvt,
    const float* __restrict__ eq_feat, const float* __restrict__ env_node,
    const int* __restrict__ ecen, const float* __restrict__ W_ls,
    const float* __restrict__ W_lv, float* out_inv, float* out_eq, int E) {
  __shared__ __align__(16) float lat[8][128];
  __shared__ __align__(16) float tpS[8][32][11];
  __shared__ __align__(16) float Wsh[64][128];   // W_env eq half (cols 0..63)
  __shared__ __align__(16) float WlsS[64 * 32];
  __shared__ __align__(16) float WlvS[96 * 32];
  const int tid = threadIdx.x;
  {
    const float4* s0 = (const float4*)Wenvt;
    float4* d0 = (float4*)&Wsh[0][0];
    for (int i = tid; i < 64 * 128 / 4; i += 256) async_copy16(d0 + i, s0 + i);
    const float4* s1 = (const float4*)W_ls;
    float4* d1 = (float4*)WlsS;
    for (int i = tid; i < 64 * 32 / 4; i += 256) async_copy16(d1 + i, s1 + i);
    const float4* s2 = (const float4*)W_lv;
    float4* d2 = (float4*)WlvS;
    for (int i = tid; i < 96 * 32 / 4; i += 256) async_copy16(d2 + i, s2 + i);
  }
  const int wave = tid >> 5, lane = tid & 31;
  const int e = blockIdx.x * 8 + wave;
  const bool ok = e < E;
  if (ok)
    for (int k = lane; k < 128; k += 32) {
      float l = latents[(size_t)e * 128 + k];
      lat[wave][k] = l;
      out_inv[(size_t)e * 192 + k] = l;     // inv_latent[:, :128]
    }
  async_join();
  __syncthreads();
  const int m = lane;
  float t[11];
  if (ok) {
    const int c = ecen[e];
    const float4* lw0 = (const float4*)&Wsh[2 * m][0];
    const float4* lw1 = (const float4*)&Wsh[2 * m + 1][0];
    const float4* ll  = (const float4*)&lat[wave][0];
    float w0 = 0.f, w1 = 0.f;
#pragma unroll 8
    for (int k = 0; k < 32; ++k) {
      float4 l = ll[k], a = lw0[k], b = lw1[k];
      w0 += l.x * a.x + l.y * a.y + l.z * a.z + l.w * a.w;
      w1 += l.x * b.x + l.y * b.y + l.z * b.z + l.w * b.w;
    }
    float f0 = eq_feat[(size_t)e * 4 + 0], f1 = eq_feat[(size_t)e * 4 + 1];
    float f2 = eq_feat[(size_t)e * 4 + 2], f3 = eq_feat[(size_t)e * 4 + 3];
    float eqs = f0 * w0, x0 = f1 * w1, x1 = f2 * w1, x2 = f3 * w1;
    const float* en = env_node + (size_t)c * 128 + m * 4;
    float e0v = en[0], y0 = en[1], y1 = en[2], y2 = en[3];
    t[0] = eqs * e0v;
    t[1] = (x0 * y0 + x1 * y1 + x2 * y2) * (1.f / SQ3);
    t[2] = eqs * y0;  t[3] = eqs * y1;  t[4] = eqs * y2;
    t[5] = x0 * e0v;  t[6] = x1 * e0v;  t[7] = x2 * e0v;
    t[8]  = (x1 * y2 - x2 * y1) * (1.f / SQ2);
    t[9]  = (x2 * y0 - x0 * y2) * (1.f / SQ2);
    t[10] = (x0 * y1 - x1 * y0) * (1.f / SQ2);
  } else {
#pragma unroll
    for (int j = 0; j < 11; ++j) t[j] = 0.f;
  }
  // SO(3) LN over blocks [0],[1],[2..4],[5..7],[8..10]
  float ss[5];
  ss[0] = t[0] * t[0];
  ss[1] = t[1] * t[1];
  ss[2] = t[2] * t[2] + t[3] * t[3] + t[4] * t[4];
  ss[3] = t[5] * t[5] + t[6] * t[6] + t[7] * t[7];
  ss[4] = t[8] * t[8] + t[9] * t[9] + t[10] * t[10];
  for (int msk = 16; msk >= 1; msk >>= 1)
#pragma unroll
    for (int b = 0; b < 5; ++b) ss[b] += __shfl_xor(ss[b], msk, 32);
  const float dims[5] = {32.f, 32.f, 96.f, 96.f, 96.f};
  float ir[5];
#pragma unroll
  for (int b = 0; b < 5; ++b) ir[b] = 1.f / sqrtf(ss[b] / dims[b] + 1e-5f);
  t[0] *= ir[0];  t[1] *= ir[1];
  t[2] *= ir[2];  t[3] *= ir[2];  t[4] *= ir[2];
  t[5] *= ir[3];  t[6] *= ir[3];  t[7] *= ir[3];
  t[8] *= ir[4];  t[9] *= ir[4];  t[10] *= ir[4];
#pragma unroll
  for (int j = 0; j < 11; ++j) tpS[wave][m][j] = t[j];
  if (ok) {
    out_inv[(size_t)e * 192 + 128 + 2 * m + 0] = t[0];
    out_inv[(size_t)e * 192 + 128 + 2 * m + 1] = t[1];
  }
  __syncthreads();
  if (!ok) return;
  float os = 0.f, ov0 = 0.f, ov1 = 0.f, ov2 = 0.f;
  for (int p = 0; p < 32; ++p) {
    const float* tpp = tpS[wave][p];
    os += tpp[0] * WlsS[p * 32 + m] + tpp[1] * WlsS[(32 + p) * 32 + m];
    float wl0 = WlvS[p * 32 + m], wl1 = WlvS[(32 + p) * 32 + m], wl2 = WlvS[(64 + p) * 32 + m];
    ov0 += tpp[2] * wl0 + tpp[5] * wl1 + tpp[8]  * wl2;
    ov1 += tpp[3] * wl0 + tpp[6] * wl1 + tpp[9]  * wl2;
    ov2 += tpp[4] * wl0 + tpp[7] * wl1 + tpp[10] * wl2;
  }
  float* d = out_eq + (size_t)e * 128 + m * 4;
  d[0] = os; d[1] = ov0; d[2] = ov1; d[3] = ov2;
}

// -------------------------------------------------------------------------
extern "C" void kernel_launch(void* const* d_in, const int* in_sizes, int n_in,
                              void* d_out, int out_size, void* d_ws, size_t ws_size,
                              hipStream_t stream) {
  (void)n_in; (void)out_size; (void)ws_size;
  const float* latents_in = (const float*)d_in[0];
  const float* inv_cat    = (const float*)d_in[1];
  const float* eq_feat    = (const float*)d_in[2];
  const float* cutoff     = (const float*)d_in[3];
  const float* edge_attr  = (const float*)d_in[4];
  const float* node_inv   = (const float*)d_in[5];
  const float* edge_inv   = (const float*)d_in[6];
  const float* ln_g = (const float*)d_in[7];
  const float* ln_b = (const float*)d_in[8];
  const float* W1   = (const float*)d_in[9];
  const float* b1   = (const float*)d_in[10];
  const float* W2   = (const float*)d_in[11];
  const float* b2   = (const float*)d_in[12];
  const float* W_env = (const float*)d_in[13];
  const float* Wq   = (const float*)d_in[14];
  const float* Wk   = (const float*)d_in[15];
  const float* W_el0 = (const float*)d_in[16];
  const float* W_el1 = (const float*)d_in[17];
  const float* W_ls = (const float*)d_in[18];
  const float* W_lv = (const float*)d_in[19];
  const int* ecen   = (const int*)d_in[20];
  const int* enb    = (const int*)d_in[21];
  const int* aedges = (const int*)d_in[22];
  const int E = in_sizes[3];          // cutoff_coeffs
  const int N = in_sizes[5] / 64;     // node_invariants

  char* ws = (char*)d_ws;
  size_t off = 0;
  auto take = [&](size_t bytes) -> char* {
    char* p = ws + off;
    off = (off + bytes + 255) & ~(size_t)255;
    return p;
  };
  _Float16* W1t = (_Float16*)take((size_t)256 * 160 * 2);
  _Float16* W2t = (_Float16*)take((size_t)128 * 256 * 2);
  _Float16* Wqt = (_Float16*)take((size_t)1024 * 160 * 2);
  _Float16* Wkt = (_Float16*)take((size_t)1024 * 128 * 2);
  float* Wenvt     = (float*)take((size_t)160 * 128 * 4);
  float* logits    = (float*)take((size_t)E * 32 * 4);
  float* mx        = (float*)take((size_t)N * 32 * 4);
  float* den       = (float*)take((size_t)N * 32 * 4);
  float* counts    = (float*)take((size_t)N * 4);
  float* local_env = (float*)take((size_t)N * 128 * 4);
  float* env_node  = (float*)take((size_t)N * 128 * 4);

  float* out     = (float*)d_out;
  float* out_lat = out;
  float* out_inv = out + (size_t)E * 128;
  float* out_eq  = out + (size_t)E * (128 + 192);

  k_w2h<<<(256 * 160 + 255) / 256, 256, 0, stream>>>(W1, W1t, 136, 256, 160);
  k_w2h<<<(128 * 256 + 255) / 256, 256, 0, stream>>>(W2, W2t, 256, 128, 256);
  k_w2h<<<(1024 * 160 + 255) / 256, 256, 0, stream>>>(Wq, Wqt, 136, 1024, 160);
  k_w2h<<<(1024 * 128 + 255) / 256, 256, 0, stream>>>(Wk, Wkt, 128, 1024, 128);
  k_w2f<<<(160 * 128 + 255) / 256, 256, 0, stream>>>(W_env, Wenvt, 128, 160);
  k_copy<<<((E * 128) + 255) / 256, 256, 0, stream>>>(latents_in, out_lat, E * 128);
  k_init<<<((N * 128) + 255) / 256, 256, 0, stream>>>(mx, den, counts, local_env, N);

  const int tiles = (E + 63) / 64;
  k_mlp<<<tiles, 256, 0, stream>>>(inv_cat, ln_g, ln_b, W1t, b1, W2t, b2,
                                   cutoff, aedges, out_lat, E);
  k_qk<<<tiles, 256, 0, stream>>>(node_inv, edge_inv, out_lat, Wqt, Wkt,
                                  ecen, enb, logits, E);
  const int EM = E * 32;
  k_segmax<<<(EM + 255) / 256, 256, 0, stream>>>(logits, ecen, mx, EM);
  k_exp<<<(EM + 255) / 256, 256, 0, stream>>>(logits, ecen, mx, den, EM);
  k_env<<<(E + 7) / 8, 256, 0, stream>>>(out_lat, Wenvt, edge_attr, logits, den,
                                         ecen, local_env, counts, E);
  k_node<<<(N + 7) / 8, 256, 0, stream>>>(local_env, counts, W_el0, W_el1,
                                          env_node, N);
  k_final<<<(E + 7) / 8, 256, 0, stream>>>(out_lat, Wenvt, eq_feat, env_node,
                                           ecen, W_ls, W_lv, out_inv, out_eq, E);
}
